// InstEncoder_15633680958207
// MI455X (gfx1250) — matrix operand order repaired and checked by reading, not compiled, not run on hardware
//
#include <hip/hip_runtime.h>

typedef __attribute__((ext_vector_type(16))) __bf16 bfvec16;
typedef __attribute__((ext_vector_type(8)))  float  fvec8;

namespace {
constexpr int kH = 64;           // hidden = input size
constexpr int kT = 128;          // timesteps
constexpr int kFragDw = 8;       // dwords per lane per 16x16x32 fragment
constexpr int kMatDwords = 16 /*j*/ * 2 /*s*/ * 32 /*lanes*/ * kFragDw; // 8192 dw
constexpr float kLog2e = 1.4426950408889634f;
}

__device__ __forceinline__ unsigned bf16_bits(float x) {
  unsigned u = __float_as_uint(x);
  return (u + 0x7FFFu + ((u >> 16) & 1u)) >> 16;   // round-to-nearest-even
}
__device__ __forceinline__ unsigned pack_bf16(float lo, float hi) {
  return bf16_bits(lo) | (bf16_bits(hi) << 16);
}
__device__ __forceinline__ float fast_rcp(float x) {
  return __builtin_amdgcn_rcpf(x);                 // single v_rcp_f32
}
__device__ __forceinline__ float exp2_hw(float x) {
  return __builtin_amdgcn_exp2f(x);                // single v_exp_f32
}
// sigmoid(x + b) with pre-scaled bias bs = -log2e * b
__device__ __forceinline__ float sigmoid_pre(float x, float bs) {
  return fast_rcp(1.0f + exp2_hw(__builtin_fmaf(x, -kLog2e, bs)));
}
// tanh(x + b) with pre-scaled bias bt = -2*log2e * b
__device__ __forceinline__ float tanh_pre(float x, float bt) {
  float e = exp2_hw(__builtin_fmaf(x, -2.0f * kLog2e, bt));
  return (1.0f - e) * fast_rcp(1.0f + e);
}
__device__ __forceinline__ float tanh_nb(float x) {
  float e = exp2_hw(x * (-2.0f * kLog2e));
  return (1.0f - e) * fast_rcp(1.0f + e);
}

union Frag {
  unsigned u[8];
  uint4    q[2];
  bfvec16  v;
};

// Rearrange W[4H=256][H=64] fp32 into WMMA B-matrix fragment order, bf16.
// Fragment (j, s): B tile K = 32s..32s+31, N = 16j..16j+15.
// B layout (32x16, 16-bit): lane l -> column n = l&15, half hf = l>>4;
// dword p holds K pair (32s + 8*hf + 2p + (p>=4 ? 8 : 0), +1): out[((j*2+s)*32 + lane)*8 + p].
__global__ void prep_weight_frags(const float* __restrict__ W,
                                  unsigned* __restrict__ out) {
  int i = blockIdx.x * blockDim.x + threadIdx.x;
  if (i >= kMatDwords) return;
  int p    = i & 7;
  int lane = (i >> 3) & 31;
  int s    = (i >> 8) & 1;
  int j    = i >> 9;
  int hf   = lane >> 4;
  int n    = 16 * j + (lane & 15);
  int k    = 32 * s + 8 * hf + 2 * p + ((p >= 4) ? 8 : 0);
  out[i] = pack_bf16(W[n * kH + k], W[n * kH + k + 1]);
}

// One-time fp32 -> bf16 conversion of the embedding table (row-major kept).
__global__ void prep_embed_bf16(const float* __restrict__ E,
                                unsigned* __restrict__ out, int nPairs) {
  int i = blockIdx.x * blockDim.x + threadIdx.x;
  if (i >= nPairs) return;
  out[i] = pack_bf16(E[2 * i], E[2 * i + 1]);
}

// One block = 16 batch rows, 4 waves. Wave w owns hidden slice [16w, 16w+16),
// i.e. gate tiles j = {w, 4+w, 8+w, 12+w} (gate order i,f,g,o along 4H).
__global__ __launch_bounds__(128, 1)
void lstm_scan_kernel(const int* __restrict__ inst,              // [B][T]
                      const unsigned short* __restrict__ embB,   // bf16 [vocab+1][H]
                      const unsigned* __restrict__ wihF,         // fragment-ordered bf16
                      const unsigned* __restrict__ whhF,
                      const float* __restrict__ b_ih,
                      const float* __restrict__ b_hh,
                      float* __restrict__ out) {                 // [B][H]
  __shared__ __align__(16) unsigned short h_lds[2][16 * kH]; // double-buffered [m][n]

  const int lane = threadIdx.x & 31;
  const int w    = threadIdx.x >> 5;   // hidden slice 0..3
  const int hf   = lane >> 4;
  const int mrow = lane & 15;
  const int b0   = blockIdx.x * 16;

  // ---- weight fragments -> registers, held across the whole scan ----
  Frag wih[4][2], whh[4][2];
#pragma unroll
  for (int g = 0; g < 4; ++g) {
#pragma unroll
    for (int s = 0; s < 2; ++s) {
      int j = 4 * g + w;
      int base = ((j * 2 + s) * 32 + lane) * kFragDw;
      wih[g][s].q[0] = *(const uint4*)(wihF + base);
      wih[g][s].q[1] = *(const uint4*)(wihF + base + 4);
      whh[g][s].q[0] = *(const uint4*)(whhF + base);
      whh[g][s].q[1] = *(const uint4*)(whhF + base + 4);
    }
  }

  // pre-scaled biases: sigmoid gates use -log2e*b, tanh gate uses -2log2e*b
  float bs0, bs1, bt2, bs3;
  {
    int n0 = 0 * 64 + 16 * w + mrow, n1 = 64 + 16 * w + mrow;
    int n2 = 128 + 16 * w + mrow,    n3 = 192 + 16 * w + mrow;
    bs0 = -kLog2e * (b_ih[n0] + b_hh[n0]);
    bs1 = -kLog2e * (b_ih[n1] + b_hh[n1]);
    bt2 = -2.0f * kLog2e * (b_ih[n2] + b_hh[n2]);
    bs3 = -kLog2e * (b_ih[n3] + b_hh[n3]);
  }

  Frag hfrag[2];
  float cstate[8], hval[8];
#pragma unroll
  for (int r = 0; r < 8; ++r) { cstate[r] = 0.0f; hval[r] = 0.0f; }
#pragma unroll
  for (int p = 0; p < 8; ++p) { hfrag[0].u[p] = 0u; hfrag[1].u[p] = 0u; }

  const fvec8 zeroC = {0.f, 0.f, 0.f, 0.f, 0.f, 0.f, 0.f, 0.f};

  const int* instRow = inst + (b0 + mrow) * kT;

  // ---- software pipeline: x fragments double-buffered, 1 iter ahead ----
  Frag xf[2][2];                       // [parity][s]
  int idx0 = instRow[0];
  {
    const unsigned short* row = embB + (long long)idx0 * kH;
    xf[0][0].q[0] = *(const uint4*)(row +      8 * hf);
    xf[0][0].q[1] = *(const uint4*)(row + 16 + 8 * hf);
    xf[0][1].q[0] = *(const uint4*)(row + 32 + 8 * hf);
    xf[0][1].q[1] = *(const uint4*)(row + 48 + 8 * hf);
  }
  int idxNext = instRow[1];

#pragma unroll 2
  for (int t = 0; t < kT; ++t) {
    const int cur = t & 1, nxt = cur ^ 1;

    // ---- prefetch x for t+1 (address chain resolved last iteration) ----
    {
      const unsigned short* row = embB + (long long)idxNext * kH;
      xf[nxt][0].q[0] = *(const uint4*)(row +      8 * hf);
      xf[nxt][0].q[1] = *(const uint4*)(row + 16 + 8 * hf);
      xf[nxt][1].q[0] = *(const uint4*)(row + 32 + 8 * hf);
      xf[nxt][1].q[1] = *(const uint4*)(row + 48 + 8 * hf);
    }
    idxNext = instRow[(t + 2 < kT) ? (t + 2) : (kT - 1)];  // idx two iters ahead

    // ---- gates = x@W_ih^T + h@W_hh^T (C starts at 0; bias folded later) ----
    fvec8 acc[4];
#pragma unroll
    for (int g = 0; g < 4; ++g) {
      acc[g] = __builtin_amdgcn_wmma_f32_16x16x32_bf16(
          false, xf[cur][0].v, false, wih[g][0].v, (short)0, zeroC,  false, false);
      acc[g] = __builtin_amdgcn_wmma_f32_16x16x32_bf16(
          false, hfrag[0].v,   false, whh[g][0].v, (short)0, acc[g], false, false);
      acc[g] = __builtin_amdgcn_wmma_f32_16x16x32_bf16(
          false, xf[cur][1].v, false, wih[g][1].v, (short)0, acc[g], false, false);
      acc[g] = __builtin_amdgcn_wmma_f32_16x16x32_bf16(
          false, hfrag[1].v,   false, whh[g][1].v, (short)0, acc[g], false, false);
    }

    // ---- elementwise LSTM cell update (C/D layout: r -> m = r+8hf) ----
#pragma unroll
    for (int r = 0; r < 8; ++r) {
      float ig = sigmoid_pre(acc[0][r], bs0);
      float fg = sigmoid_pre(acc[1][r], bs1);
      float gg = tanh_pre(acc[2][r], bt2);
      float og = sigmoid_pre(acc[3][r], bs3);
      float c  = fg * cstate[r] + ig * gg;
      cstate[r] = c;
      hval[r]   = og * tanh_nb(c);
    }

    // ---- relayout h via double-buffered LDS: single barrier per step ----
#pragma unroll
    for (int r = 0; r < 8; ++r) {
      int m = r + 8 * hf;
      h_lds[cur][m * kH + 16 * w + mrow] = (unsigned short)bf16_bits(hval[r]);
    }
    __syncthreads();   // RAW for this step; WAR vs step t+2 separated by t+1's barrier
#pragma unroll
    for (int s = 0; s < 2; ++s) {
      hfrag[s].q[0] = *(const uint4*)&h_lds[cur][mrow * kH + 32 * s + 8 * hf];
      hfrag[s].q[1] = *(const uint4*)&h_lds[cur][mrow * kH + 32 * s + 16 + 8 * hf];
    }
  }

  // ---- final h -> out[B][H] ----
#pragma unroll
  for (int r = 0; r < 8; ++r) {
    int m = r + 8 * hf;
    out[(b0 + m) * kH + 16 * w + mrow] = hval[r];
  }
}

extern "C" void kernel_launch(void* const* d_in, const int* in_sizes, int n_in,
                              void* d_out, int out_size, void* d_ws, size_t ws_size,
                              hipStream_t stream) {
  const int*   inst  = (const int*)  d_in[0];  // [B][T]
  const float* embed = (const float*)d_in[1];  // [vocab+1][H]
  const float* W_ih  = (const float*)d_in[2];  // [4H][H]
  const float* W_hh  = (const float*)d_in[3];  // [4H][H]
  const float* b_ih  = (const float*)d_in[4];  // [4H]
  const float* b_hh  = (const float*)d_in[5];  // [4H]
  float* out = (float*)d_out;                  // [B][H]

  unsigned* wihF = (unsigned*)d_ws;
  unsigned* whhF = wihF + kMatDwords;                    // +32KB
  unsigned* embB = whhF + kMatDwords;                    // +32KB, 16B aligned

  dim3 pb(256), pg((kMatDwords + 255) / 256);
  prep_weight_frags<<<pg, pb, 0, stream>>>(W_ih, wihF);
  prep_weight_frags<<<pg, pb, 0, stream>>>(W_hh, whhF);

  int nPairs = in_sizes[1] / 2;                          // (vocab+1)*H / 2 dwords
  prep_embed_bf16<<<(nPairs + 255) / 256, 256, 0, stream>>>(embed, embB, nPairs);

  int B = in_sizes[0] / kT;                              // 4096
  lstm_scan_kernel<<<B / 16, 128, 0, stream>>>(
      inst, (const unsigned short*)embB, wihF, whhF, b_ih, b_hh, out);
}
